// Decoder_40604620816467
// MI455X (gfx1250) — matrix operand order repaired and checked by reading, not compiled, not run on hardware
//
#include <hip/hip_runtime.h>
#include <stddef.h>
#include <stdint.h>

// ---------------- constants (match reference) ----------------
#define BB   256
#define LL   2
#define HH   1024
#define EE   256
#define VV   28
#define ZZ   1024
#define TT   512
#define SOS_IDX 1

#define PANEL 128                 // K-panel staged in LDS
#define LDSB  (3 * 16 * PANEL)    // halves per buffer (3 gates x 16 cols x PANEL)

// ---------------- WMMA vector types ----------------
typedef __attribute__((ext_vector_type(16))) __bf16 v16bf;
typedef __attribute__((ext_vector_type(8)))  float  v8f;

union ABu { v16bf v; unsigned int u[8]; uint4 q[2]; };

// ---------------- helpers ----------------
__device__ __forceinline__ float bf2f(unsigned short h) {
    unsigned int u = ((unsigned int)h) << 16;
    float f;
    __builtin_memcpy(&f, &u, 4);
    return f;
}

__device__ __forceinline__ unsigned short f2bf(float f) {
    unsigned int u;
    __builtin_memcpy(&u, &f, 4);
    if ((u & 0x7fffffffu) > 0x7f800000u) return (unsigned short)((u >> 16) | 0x40); // quiet NaN
    unsigned int r = u + 0x7fffu + ((u >> 16) & 1u); // round-to-nearest-even
    return (unsigned short)(r >> 16);
}

__device__ __forceinline__ float gumbel_hash(unsigned long long x) {
    x += 0x9E3779B97F4A7C15ull;
    x = (x ^ (x >> 30)) * 0xBF58476D1CE4E5B9ull;
    x = (x ^ (x >> 27)) * 0x94D049BB133111EBull;
    x ^= x >> 31;
    float u = (float)((x >> 40) + 1ull) * (1.0f / 16777217.0f);
    return -__logf(-__logf(u));
}

// CDNA5 async global->LDS copy, 16B per lane, tracked by ASYNCcnt.
// LDS generic pointers carry the LDS byte offset in their low 32 bits.
__device__ __forceinline__ void async_copy16(unsigned lds_off, const void* gaddr) {
    asm volatile("global_load_async_to_lds_b128 %0, %1, off"
                 :: "v"(lds_off), "v"(gaddr)
                 : "memory");
}
__device__ __forceinline__ void wait_asynccnt0() {
    asm volatile("s_wait_asynccnt 0" ::: "memory");
}

// ---------------- setup kernels ----------------

__global__ void cvt_bf16_kernel(const float* __restrict__ src,
                                unsigned short* __restrict__ dst, int n) {
    for (int i = blockIdx.x * blockDim.x + threadIdx.x; i < n;
         i += gridDim.x * blockDim.x) {
        dst[i] = f2bf(src[i]);
    }
}

// h = tanh(z @ w_zh^T + b_zh), split into h0/h1 (fp32 + bf16 copies)
__global__ void init_h_kernel(const float* __restrict__ z,
                              const float* __restrict__ wzh,
                              const float* __restrict__ bzh,
                              float* __restrict__ h0fp, unsigned short* __restrict__ h0bf,
                              float* __restrict__ h1fp, unsigned short* __restrict__ h1bf) {
    int idx = blockIdx.x * blockDim.x + threadIdx.x; // over B * (L*H)
    if (idx >= BB * LL * HH) return;
    int b = idx >> 11;
    int r = idx & (LL * HH - 1);
    const float* zr = z + (size_t)b * ZZ;
    const float* wr = wzh + (size_t)r * ZZ;
    float acc = 0.f;
    #pragma unroll 4
    for (int k = 0; k < ZZ; ++k) acc = fmaf(zr[k], wr[k], acc);
    float v = tanhf(acc + bzh[r]);
    int l = r >> 10;
    int j = r & (HH - 1);
    size_t o = (size_t)b * HH + j;
    if (l == 0) { h0fp[o] = v; h0bf[o] = f2bf(v); }
    else        { h1fp[o] = v; h1bf[o] = f2bf(v); }
}

__global__ void init_tok_kernel(int* __restrict__ tok) {
    int i = blockIdx.x * blockDim.x + threadIdx.x;
    if (i < BB) tok[i] = SOS_IDX;
}

// ---------------- staged 3-gate GEMM helper ----------------
// All 4 waves of the block share one column tile (col0): the 3-gate B panel
// (3 x 16 x PANEL bf16) is filled by async global->LDS copies (double-
// buffered, zero VGPR staging) and consumed via ds_load_b128 by every wave;
// A fragments are per-wave global_load_b128 pairs.
__device__ __forceinline__ void gemm3_staged(
    const unsigned short* __restrict__ arow,  // per-lane A row base (stride Ks)
    const unsigned short* __restrict__ wgt,   // (3H, Ks) bf16
    int Ks, unsigned col0,
    unsigned tid, unsigned half, unsigned lm,
    unsigned short* lds,                      // 2 * LDSB halves
    v8f& acc0, v8f& acc1, v8f& acc2)
{
    const int P = Ks / PANEL;

    // cooperative async fill of one panel: 6 x 16B chunks per thread
    auto issueAsync = [&](int p, int buf) {
        unsigned short* base = lds + buf * LDSB;
        #pragma unroll
        for (int i = 0; i < 6; ++i) {
            int idx = (int)tid + 128 * i;       // 0..767
            int g   = idx >> 8;                 // gate
            int rem = idx & 255;
            int n   = rem >> 4;                 // column within tile
            int c   = rem & 15;                 // 16-byte chunk within panel row
            const unsigned short* src =
                wgt + (size_t)(g * HH + col0 + n) * (size_t)Ks + p * PANEL + c * 8;
            unsigned dst = (unsigned)(uintptr_t)(base + (g * 16 + n) * PANEL + c * 8);
            async_copy16(dst, src);
        }
    };

    issueAsync(0, 0);
    wait_asynccnt0();
    __syncthreads();

    for (int p = 0; p < P; ++p) {
        if (p + 1 < P) issueAsync(p + 1, (p + 1) & 1);   // overlap with compute
        const unsigned short* base = lds + (p & 1) * LDSB;
        #pragma unroll
        for (int k0 = 0; k0 < PANEL; k0 += 32) {
            ABu a, b;
            // ISA 16-bit A layout: u[0..3] = halves[k0+half*8 ..], u[4..7] = +16
            const unsigned short* ap = arow + p * PANEL + k0 + half * 8;
            a.q[0] = *(const uint4*)(ap);
            a.q[1] = *(const uint4*)(ap + 16);
            // B layout matches A: u[0..3] = halves[k0+half*8 ..], u[4..7] = +16
            const unsigned short* bp0 = base + (0 * 16 + lm) * PANEL + k0 + half * 8;
            b.q[0] = *(const uint4*)(bp0);
            b.q[1] = *(const uint4*)(bp0 + 16);
            acc0 = __builtin_amdgcn_wmma_f32_16x16x32_bf16(false, a.v, false, b.v,
                                                           (short)0, acc0, false, false);
            const unsigned short* bp1 = base + (1 * 16 + lm) * PANEL + k0 + half * 8;
            b.q[0] = *(const uint4*)(bp1);
            b.q[1] = *(const uint4*)(bp1 + 16);
            acc1 = __builtin_amdgcn_wmma_f32_16x16x32_bf16(false, a.v, false, b.v,
                                                           (short)0, acc1, false, false);
            const unsigned short* bp2 = base + (2 * 16 + lm) * PANEL + k0 + half * 8;
            b.q[0] = *(const uint4*)(bp2);
            b.q[1] = *(const uint4*)(bp2 + 16);
            acc2 = __builtin_amdgcn_wmma_f32_16x16x32_bf16(false, a.v, false, b.v,
                                                           (short)0, acc2, false, false);
        }
        if (p + 1 < P) { wait_asynccnt0(); __syncthreads(); }
    }
    __syncthreads();   // LDS is reused by the caller's next phase
}

// ---------------- fused GRU cell (one kernel per layer per step) ----------------
// Block = 4 waves sharing one 16-column tile (B panels async-staged in LDS),
// each wave owns one 16-row batch tile. Six f32 WMMA accumulators per wave;
// sigmoid/tanh gate math fused in the epilogue.
__global__ __launch_bounds__(128)
void gru_cell_kernel(const int* __restrict__ tok,
                     const unsigned short* __restrict__ embA, // gather table (layer0)
                     int useGather, int Kx,
                     const unsigned short* __restrict__ xbf,  // direct x (layer1)
                     const unsigned short* __restrict__ wih,  // (3H, Kx) bf16
                     const unsigned short* __restrict__ whh,  // (3H, H) bf16
                     const float* __restrict__ bih,
                     const float* __restrict__ bhh,
                     const float* __restrict__ hprev_fp,
                     const unsigned short* __restrict__ hprev_bf,
                     float* __restrict__ hnext_fp,
                     unsigned short* __restrict__ hnext_bf) {
    const int H = HH;
    __shared__ unsigned short ldsB[2 * LDSB];   // 24 KB

    unsigned tid  = threadIdx.x;
    unsigned lane = tid & 31u;
    unsigned wv   = tid >> 5;
    unsigned colTile = blockIdx.x & 63u;             // shared by the 4 waves
    unsigned rowTile = (blockIdx.x >> 6) * 4u + wv;  // unique per wave (0..15)
    unsigned row0 = rowTile * 16u;
    unsigned col0 = colTile * 16u;
    unsigned half = lane >> 4;
    unsigned lm   = lane & 15u;

    v8f accI0 = {}, accI1 = {}, accI2 = {};
    v8f accH0 = {}, accH1 = {}, accH2 = {};

    // ---- x-GEMM (layer0: gathered through token) ----
    const unsigned short* arow =
        useGather ? embA + (size_t)tok[row0 + lm] * (size_t)Kx
                  : xbf  + (size_t)(row0 + lm)    * (size_t)Kx;
    gemm3_staged(arow, wih, Kx, col0, tid, half, lm, ldsB, accI0, accI1, accI2);

    // ---- h-GEMM ----
    const unsigned short* hrow = hprev_bf + (size_t)(row0 + lm) * H;
    gemm3_staged(hrow, whh, H, col0, tid, half, lm, ldsB, accH0, accH1, accH2);

    // ---- fused gate epilogue ----
    unsigned ncol = col0 + lm;
    float bir = bih[ncol], biz = bih[H + ncol], bin = bih[2 * H + ncol];
    float bhr = bhh[ncol], bhz = bhh[H + ncol], bhn = bhh[2 * H + ncol];
    #pragma unroll
    for (int r = 0; r < 8; ++r) {
        unsigned m = row0 + r + (half << 3);   // ISA C/D layout: lanes16-31 -> M+8
        float ir = accI0[r] + bir, iz = accI1[r] + biz, in_ = accI2[r] + bin;
        float hr = accH0[r] + bhr, hz = accH1[r] + bhz, hn  = accH2[r] + bhn;
        float rg = 1.f / (1.f + __expf(-(ir + hr)));
        float zg = 1.f / (1.f + __expf(-(iz + hz)));
        float ng = tanhf(in_ + rg * hn);
        float hp = hprev_fp[(size_t)m * H + ncol];
        float hv = (1.f - zg) * ng + zg * hp;
        hnext_fp[(size_t)m * H + ncol] = hv;
        hnext_bf[(size_t)m * H + ncol] = f2bf(hv);
    }
}

// ---------------- logits + Gumbel-max sampling (one wave per batch row) ----------------
__global__ __launch_bounds__(32)
void logits_sample_kernel(const float* __restrict__ h1fp,
                          const unsigned short* __restrict__ woutbf, // (V, H) bf16
                          const float* __restrict__ bout,
                          float* __restrict__ out_logits,  // (B, T, V)
                          float* __restrict__ out_tokens,  // (B, T)
                          int* __restrict__ tok, int t) {
    int b = blockIdx.x;
    int v = threadIdx.x;
    float logit = -__builtin_inff();
    if (v < VV) {
        const float* h = h1fp + (size_t)b * HH;
        const unsigned short* w = woutbf + (size_t)v * HH;
        float acc = 0.f;
        #pragma unroll 4
        for (int k = 0; k < HH; ++k) acc = fmaf(h[k], bf2f(w[k]), acc);
        logit = acc + bout[v];
        out_logits[((size_t)b * TT + t) * VV + v] = logit;
    }
    float key = -__builtin_inff();
    if (v < VV) {
        unsigned long long s =
            ((unsigned long long)t << 20) | ((unsigned long long)b << 6) | (unsigned)v;
        key = logit + gumbel_hash(s);
    }
    int best = v;
    float bk = key;
    #pragma unroll
    for (int off = 16; off >= 1; off >>= 1) {
        float ok = __shfl_xor(bk, off, 32);
        int   ob = __shfl_xor(best, off, 32);
        if (ok > bk || (ok == bk && ob < best)) { bk = ok; best = ob; }
    }
    if (v == 0) {
        tok[b] = best;
        out_tokens[(size_t)b * TT + t] = (float)best;
    }
}

// ---------------- host-side orchestration ----------------
extern "C" void kernel_launch(void* const* d_in, const int* in_sizes, int n_in,
                              void* d_out, int out_size, void* d_ws, size_t ws_size,
                              hipStream_t stream) {
    const float* z    = (const float*)d_in[0];
    const float* emb  = (const float*)d_in[1];
    const float* wzh  = (const float*)d_in[2];
    const float* bzh  = (const float*)d_in[3];
    const float* wih0 = (const float*)d_in[4];
    const float* whh0 = (const float*)d_in[5];
    const float* bih0 = (const float*)d_in[6];
    const float* bhh0 = (const float*)d_in[7];
    const float* wih1 = (const float*)d_in[8];
    const float* whh1 = (const float*)d_in[9];
    const float* bih1 = (const float*)d_in[10];
    const float* bhh1 = (const float*)d_in[11];
    const float* wout = (const float*)d_in[12];
    const float* bout = (const float*)d_in[13];
    (void)in_sizes; (void)n_in; (void)out_size; (void)ws_size;

    // ---- workspace carve-up ----
    char* ws = (char*)d_ws;
    size_t off = 0;
    auto carve = [&](size_t bytes) -> char* {
        char* p = ws + off;
        off += (bytes + 255) & ~(size_t)255;
        return p;
    };
    unsigned short* emb_bf  = (unsigned short*)carve((size_t)VV * EE * 2);
    unsigned short* wih0_bf = (unsigned short*)carve((size_t)3 * HH * EE * 2);
    unsigned short* whh0_bf = (unsigned short*)carve((size_t)3 * HH * HH * 2);
    unsigned short* wih1_bf = (unsigned short*)carve((size_t)3 * HH * HH * 2);
    unsigned short* whh1_bf = (unsigned short*)carve((size_t)3 * HH * HH * 2);
    unsigned short* wout_bf = (unsigned short*)carve((size_t)VV * HH * 2);
    float* h0fp[2]; float* h1fp[2];
    unsigned short* h0bf[2]; unsigned short* h1bf[2];
    for (int i = 0; i < 2; ++i) {
        h0fp[i] = (float*)carve((size_t)BB * HH * 4);
        h1fp[i] = (float*)carve((size_t)BB * HH * 4);
        h0bf[i] = (unsigned short*)carve((size_t)BB * HH * 2);
        h1bf[i] = (unsigned short*)carve((size_t)BB * HH * 2);
    }
    int* tok = (int*)carve((size_t)BB * 4);

    float* out_logits = (float*)d_out;                     // (B, T, V)
    float* out_tokens = out_logits + (size_t)BB * TT * VV; // (B, T)

    // ---- setup: bf16 conversions (every call; deterministic) ----
    auto cvt = [&](const float* s, unsigned short* d, int n) {
        int grid = (n + 255) / 256;
        if (grid > 4096) grid = 4096;
        cvt_bf16_kernel<<<grid, 256, 0, stream>>>(s, d, n);
    };
    cvt(emb,  emb_bf,  VV * EE);
    cvt(wih0, wih0_bf, 3 * HH * EE);
    cvt(whh0, whh0_bf, 3 * HH * HH);
    cvt(wih1, wih1_bf, 3 * HH * HH);
    cvt(whh1, whh1_bf, 3 * HH * HH);
    cvt(wout, wout_bf, VV * HH);

    init_h_kernel<<<(BB * LL * HH + 255) / 256, 256, 0, stream>>>(
        z, wzh, bzh, h0fp[0], h0bf[0], h1fp[0], h1bf[0]);
    init_tok_kernel<<<1, BB, 0, stream>>>(tok);

    // ---- sequential decode: 3 fused kernels per step ----
    for (int t = 0; t < TT; ++t) {
        int p = t & 1, q = p ^ 1;
        // layer 0: x = emb[tok] (gathered), K = E
        gru_cell_kernel<<<256, 128, 0, stream>>>(
            tok, emb_bf, /*useGather=*/1, /*Kx=*/EE, /*xbf=*/nullptr,
            wih0_bf, whh0_bf, bih0, bhh0,
            h0fp[p], h0bf[p], h0fp[q], h0bf[q]);
        // layer 1: x = h0_next, K = H
        gru_cell_kernel<<<256, 128, 0, stream>>>(
            nullptr, nullptr, /*useGather=*/0, /*Kx=*/HH, /*xbf=*/h0bf[q],
            wih1_bf, whh1_bf, bih1, bhh1,
            h1fp[p], h1bf[p], h1fp[q], h1bf[q]);
        // logits + sample next token
        logits_sample_kernel<<<BB, 32, 0, stream>>>(
            h1fp[q], wout_bf, bout, out_logits, out_tokens, tok, t);
    }
}
